// DRMMTKS_class_80625126081184
// MI455X (gfx1250) — matrix operand order repaired and checked by reading, not compiled, not run on hardware
//
#include <hip/hip_runtime.h>

#define VOCAB 50000
#define EMB   300
#define KPAD  320      // 300 padded to 10 x 32 for the bf16 WMMA K-loop
#define BATCH 128
#define TQ    32
#define TD    4096
#define TOPK  20
#define NCHUNK 128     // doc tokens per workgroup in the GEMM kernel

typedef __attribute__((ext_vector_type(16))) __bf16       v16bf;
typedef __attribute__((ext_vector_type(8)))  float        v8f;
typedef __attribute__((ext_vector_type(4)))  unsigned int u32x4;
struct U4x2 { u32x4 lo, hi; };

__device__ __forceinline__ unsigned short f2bf(float f) {
  unsigned int u = __float_as_uint(f);
  u += 0x7fffu + ((u >> 16) & 1u);          // round-to-nearest-even
  return (unsigned short)(u >> 16);
}

// ---------------------------------------------------------------------------
// Kernel 1: normalize every vocab row once, emit bf16 rows padded to KPAD.
// One wave (32 lanes) per row; 8 rows per 256-thread block.
// The 60 MB fp32 table / 32 MB bf16 result both stay L2-resident (192 MB).
// ---------------------------------------------------------------------------
__global__ void normalize_emb_kernel(const float* __restrict__ emb,
                                     unsigned short* __restrict__ ntab) {
  int wave = threadIdx.x >> 5;
  int lane = threadIdx.x & 31;
  int row  = blockIdx.x * 8 + wave;          // grid sized so row < VOCAB exactly
  const float* src = emb + (size_t)row * EMB;

  float s = 0.f;
  for (int k = lane; k < EMB; k += 32) { float v = src[k]; s += v * v; }
#pragma unroll
  for (int m = 16; m >= 1; m >>= 1) s += __shfl_xor(s, m, 32);
  float inv = 1.f / fmaxf(sqrtf(s), 1e-8f);

  unsigned short* dst = ntab + (size_t)row * KPAD;
  for (int k = lane; k < KPAD; k += 32)
    dst[k] = (k < EMB) ? f2bf(src[k] * inv) : (unsigned short)0;
}

// ---------------------------------------------------------------------------
// Kernel 2: cos_mat tiles via V_WMMA_F32_16X16X32_BF16, register-blocked.
// Grid: (TD/NCHUNK, BATCH). Block: 256 threads = 8 waves. Dynamic LDS:
//   ldsQ  32 x 320 bf16 (20 KB)  +  ldsD 128 x 320 bf16 (80 KB) = 100 KB.
// 16 output tiles per WG (2 mtiles x 8 ntiles); each wave owns 2 tiles that
// share the same A fragment -> 3 ds_load_b128 per WMMA instead of 4, and two
// independent WMMAs per k-step for ILP. Zero divergence -> EXEC all-ones.
// ---------------------------------------------------------------------------
__global__ void cosine_gemm_kernel(const unsigned short* __restrict__ ntab,
                                   const int* __restrict__ query,
                                   const int* __restrict__ doc,
                                   float* __restrict__ cosbuf) {
  extern __shared__ __align__(16) unsigned short lds[];
  unsigned short* ldsQ = lds;                 // TQ * KPAD
  unsigned short* ldsD = lds + TQ * KPAD;     // NCHUNK * KPAD

  int b      = blockIdx.y;
  int nchunk = blockIdx.x;       // NCHUNK doc tokens per block
  int tid    = threadIdx.x;

  // Gather normalized bf16 rows into LDS (uniform trip counts -> no divergence).
  u32x4* q4 = (u32x4*)ldsQ;
  for (int idx = tid; idx < TQ * (KPAD / 8); idx += 256) {        // 1280 = 5*256
    int r = idx / (KPAD / 8), c = idx % (KPAD / 8);
    int tok = query[b * TQ + r];
    q4[idx] = *((const u32x4*)(ntab + (size_t)tok * KPAD) + c);
  }
  u32x4* d4 = (u32x4*)ldsD;
  for (int idx = tid; idx < NCHUNK * (KPAD / 8); idx += 256) {    // 5120 = 20*256
    int r = idx / (KPAD / 8), c = idx % (KPAD / 8);
    int tok = doc[b * TD + nchunk * NCHUNK + r];
    d4[idx] = *((const u32x4*)(ntab + (size_t)tok * KPAD) + c);
  }
  __syncthreads();

  int wave = tid >> 5, lane = tid & 31;
  int mtile = wave & 1;          // which 16 query rows
  int npair = wave >> 1;         // which pair of 16-token column tiles (0..3)

  // A frag (16x32 bf16): lane row = lane&15; 16 contiguous K at (lane>>4)*16.
  const unsigned short* qbase =
      ldsQ + (size_t)(mtile * 16 + (lane & 15)) * KPAD + ((lane >> 4) * 16);
  // B frag (32x16 bf16): lane col = lane&15; 16 contiguous K at (lane>>4)*16.
  const unsigned short* dbase0 =
      ldsD + (size_t)(npair * 32 + (lane & 15)) * KPAD + ((lane >> 4) * 16);
  const unsigned short* dbase1 = dbase0 + (size_t)16 * KPAD;

  v8f acc0 = {0.f, 0.f, 0.f, 0.f, 0.f, 0.f, 0.f, 0.f};
  v8f acc1 = {0.f, 0.f, 0.f, 0.f, 0.f, 0.f, 0.f, 0.f};
#pragma unroll
  for (int kk = 0; kk < KPAD / 32; ++kk) {
    int ko = kk * 32;
    U4x2 ua, ub0, ub1;
    ua.lo  = *(const u32x4*)(qbase + ko);
    ua.hi  = *(const u32x4*)(qbase + ko + 8);
    ub0.lo = *(const u32x4*)(dbase0 + ko);
    ub0.hi = *(const u32x4*)(dbase0 + ko + 8);
    ub1.lo = *(const u32x4*)(dbase1 + ko);
    ub1.hi = *(const u32x4*)(dbase1 + ko + 8);
    v16bf af  = __builtin_bit_cast(v16bf, ua);
    v16bf bf0 = __builtin_bit_cast(v16bf, ub0);
    v16bf bf1 = __builtin_bit_cast(v16bf, ub1);
    acc0 = __builtin_amdgcn_wmma_f32_16x16x32_bf16(
        false, af, false, bf0, (short)0, acc0, false, false);
    acc1 = __builtin_amdgcn_wmma_f32_16x16x32_bf16(
        false, af, false, bf1, (short)0, acc1, false, false);
  }

  // C/D layout: lane n = lane&15, rows m = (lane>>4)*8 + i.
  int n = lane & 15, mb = (lane >> 4) * 8;
  size_t base = ((size_t)b * TQ + mtile * 16 + mb) * TD
              + nchunk * NCHUNK + npair * 32 + n;
#pragma unroll
  for (int i = 0; i < 8; ++i) {
    cosbuf[base + (size_t)i * TD]      = acc0[i];
    cosbuf[base + (size_t)i * TD + 16] = acc1[i];
  }
}

// ---------------------------------------------------------------------------
// Kernel 3: top-20 per query row + FFW/tanh + idf-gated softmax + output.
// One block per batch, 128 threads: 4 lanes per query row.
// ---------------------------------------------------------------------------
__device__ __forceinline__ void tk_insert(float (&t)[TOPK], float v) {
  if (v > t[0]) {                 // t kept sorted ascending, t[0] == min
    t[0] = v;
#pragma unroll
    for (int i = 0; i < TOPK - 1; ++i) {
      if (t[i] > t[i + 1]) { float tmp = t[i]; t[i] = t[i + 1]; t[i + 1] = tmp; }
    }
  }
}

__global__ void topk_score_kernel(const float* __restrict__ cosbuf,
                                  const int* __restrict__ query,
                                  const float* __restrict__ idf,
                                  const float* __restrict__ ffw_W,
                                  const float* __restrict__ ffw_b,
                                  const float* __restrict__ gates_W,
                                  const float* __restrict__ out_W,
                                  const float* __restrict__ out_b,
                                  float* __restrict__ out) {
  __shared__ float cand[TQ][4 * TOPK];
  __shared__ float ffw_lds[TQ];

  int b = blockIdx.x;
  int tid = threadIdx.x;               // 128
  int q = tid >> 2, sub = tid & 3;

  const float4* row = (const float4*)(cosbuf + ((size_t)b * TQ + q) * TD);
  float t[TOPK];
#pragma unroll
  for (int i = 0; i < TOPK; ++i) t[i] = -1e30f;

  for (int c = sub; c < TD / 4; c += 4) {
    float4 v = row[c];
    tk_insert(t, v.x); tk_insert(t, v.y); tk_insert(t, v.z); tk_insert(t, v.w);
  }
#pragma unroll
  for (int i = 0; i < TOPK; ++i) cand[q][sub * TOPK + i] = t[i];
  __syncthreads();

  if (sub == 0) {
    for (int j = TOPK; j < 4 * TOPK; ++j) tk_insert(t, cand[q][j]);
    // top_k is sorted descending in the reference; t is ascending.
    float s = ffw_b[0];
#pragma unroll
    for (int j = 0; j < TOPK; ++j) s += t[TOPK - 1 - j] * ffw_W[j];
    ffw_lds[q] = tanhf(s);
  }
  __syncthreads();

  if (tid < TQ) {                       // exactly wave 0, all 32 lanes active
    int tok = query[b * TQ + tid];
    float logit = idf[b * TQ + tid] * gates_W[0] + ((tok != 0) ? 0.f : -1e7f);
    float m = logit;
#pragma unroll
    for (int d = 16; d >= 1; d >>= 1) m = fmaxf(m, __shfl_xor(m, d, 32));
    float e = expf(logit - m);
    float se = e;
#pragma unroll
    for (int d = 16; d >= 1; d >>= 1) se += __shfl_xor(se, d, 32);
    float contrib = (e / se) * ffw_lds[tid];
#pragma unroll
    for (int d = 16; d >= 1; d >>= 1) contrib += __shfl_xor(contrib, d, 32);
    if (tid == 0) out[b] = contrib * out_W[0] + out_b[0];
  }
}

// ---------------------------------------------------------------------------
extern "C" void kernel_launch(void* const* d_in, const int* in_sizes, int n_in,
                              void* d_out, int out_size, void* d_ws, size_t ws_size,
                              hipStream_t stream) {
  const int*   doc    = (const int*)d_in[0];
  const int*   query  = (const int*)d_in[1];
  const float* idf    = (const float*)d_in[2];
  const float* emb    = (const float*)d_in[3];
  const float* ffw_W  = (const float*)d_in[4];
  const float* ffw_b  = (const float*)d_in[5];
  const float* gatesW = (const float*)d_in[6];
  const float* out_W  = (const float*)d_in[7];
  const float* out_b  = (const float*)d_in[8];
  (void)in_sizes; (void)n_in; (void)out_size; (void)ws_size;

  unsigned short* ntab = (unsigned short*)d_ws;                     // 32 MB
  float* cosbuf = (float*)((char*)d_ws +
                           (size_t)VOCAB * KPAD * sizeof(unsigned short)); // 64 MB

  size_t gemm_lds = (size_t)(TQ + NCHUNK) * KPAD * sizeof(unsigned short); // 100 KB

  normalize_emb_kernel<<<VOCAB / 8, 256, 0, stream>>>(emb, ntab);
  cosine_gemm_kernel<<<dim3(TD / NCHUNK, BATCH), 256, gemm_lds, stream>>>(
      ntab, query, doc, cosbuf);
  topk_score_kernel<<<BATCH, 128, 0, stream>>>(cosbuf, query, idf, ffw_W, ffw_b,
                                               gatesW, out_W, out_b, (float*)d_out);
}
